// RoleEmergentAgent_50173807952050
// MI455X (gfx1250) — compile-verified, matrix operand order checked by reading, not run on hardware
//
#include <hip/hip_runtime.h>
#include <hip/hip_bf16.h>

// ---------------------------------------------------------------------------
// RoleEmergentAgent forward for MI455X (gfx1250, wave32, WMMA bf16).
// GEMMs on v_wmma_f32_16x16x32_bf16 (f32 accumulate), 128-row macro-tiles
// (8 WMMAs per B-fragment load), async global->LDS staging (ASYNCcnt),
// fused entity embed+LN+k/v, VALU attention core (N<=10, dh=64).
// ---------------------------------------------------------------------------

typedef __bf16 bf16_t;
typedef __attribute__((ext_vector_type(16))) __bf16 v16bf;
typedef __attribute__((ext_vector_type(8)))  float  v8f;

#define LDSX 36   // padded row stride (floats) of staged X tiles (144B: 16B-aligned)
#define LNS  264  // padded row stride (floats) of 16x256 LN buffer
#define GEMM_MT 8 // M-subtiles (of 16 rows) per k_gemm block => 128 rows

// Use CDNA5 async global->LDS copies (ASYNCcnt) for k_gemm staging.
#define ASYNC_STAGE 1

// ---- WMMA fragment helpers (layouts per CDNA5 ISA 7.12.2) -----------------

// A fragment (16x32 bf16) from an f32 LDS tile, row-major with given stride.
static __device__ inline v16bf load_a_lds(const float* base, int stride,
                                          int k0, int lane) {
  const int row = lane & 15;
  const int kg  = (lane >> 4) & 1;
  const float* p = base + row * stride + k0;
  v16bf a;
#pragma unroll
  for (int h = 0; h < 8; ++h) a[h]     = (__bf16)p[kg * 8 + h];
#pragma unroll
  for (int h = 0; h < 8; ++h) a[8 + h] = (__bf16)p[16 + kg * 8 + h];
  return a;
}

// B fragment (32x16 bf16) from a global f32 weight matrix.
// TRANSB=false: W is (K,N) row-major.  TRANSB=true: W is (N,K) row-major.
template <bool TRANSB>
static __device__ inline v16bf load_b_glb(const float* __restrict__ W,
                                          int K, int N, int k0, int nbase,
                                          int lane) {
  const int col = lane & 15;
  const int kg  = (lane >> 4) & 1;
  const int n   = nbase + col;
  v16bf b;
#pragma unroll
  for (int h = 0; h < 16; ++h) {
    const int k = k0 + ((h < 8) ? (kg * 8 + h) : (16 + kg * 8 + (h - 8)));
    float w = 0.f;
    if (k < K && n < N)
      w = TRANSB ? W[(size_t)n * K + k] : W[(size_t)k * N + n];
    b[h] = (__bf16)w;
  }
  return b;
}

static __device__ inline v8f wmma_bf16(v16bf a, v16bf b, v8f c) {
  return __builtin_amdgcn_wmma_f32_16x16x32_bf16(
      /*neg_a=*/false, a, /*neg_b=*/false, b,
      /*c_mod=*/(short)0, c, /*reuse_a=*/false, /*reuse_b=*/false);
}

// ---------------------------------------------------------------------------
// Generic GEMM: Y(M,N) = act(X(M,K) @ W + bias), 128 rows x 64 cols per block.
// Block: 128 threads = 4 waves; wave w owns N-tile blockIdx.y*64 + w*16 and
// runs 8 WMMAs (one per 16-row M-subtile) per B-fragment load.
// ---------------------------------------------------------------------------
template <bool TRANSB, int ACT>  // ACT: 0 = none, 1 = relu
__global__ void k_gemm(const float* __restrict__ X, int ldx, int xoff,
                       const float* __restrict__ W,
                       const float* __restrict__ bias,
                       float* __restrict__ Y, int M, int N, int K) {
  __shared__ float xs[128 * LDSX];
  const int tid  = threadIdx.x;
  const int lane = tid & 31;
  const int wave = tid >> 5;
  const int m0   = blockIdx.x * 128;
  const int nb   = blockIdx.y * 64 + wave * 16;

  v8f c[GEMM_MT];
#pragma unroll
  for (int t = 0; t < GEMM_MT; ++t) c[t] = (v8f){};

  for (int k0 = 0; k0 < K; k0 += 32) {
    // ---- stage 128x32 f32 tile: one float4 (16B) per item ----
    for (int i = tid; i < 128 * 8; i += blockDim.x) {
      const int r = i >> 3, q4 = i & 7;
      const int gm = m0 + r, gk = k0 + q4 * 4;
      const unsigned lds_off = (unsigned)(r * LDSX + q4 * 4) * 4u;  // xs @ LDS base
      if (gm < M && gk + 3 < K) {
        const float* src = X + (size_t)gm * ldx + xoff + gk;
#if ASYNC_STAGE
        // GLOBAL_LOAD_ASYNC_TO_LDS_B128: LDS[vdst] = MEM[vaddr], ASYNCcnt.
        asm volatile("global_load_async_to_lds_b128 %0, %1, off"
                     :: "v"(lds_off), "v"((unsigned long long)src)
                     : "memory");
#else
        *(float4*)(&xs[r * LDSX + q4 * 4]) = *(const float4*)src;
#endif
      } else {
        float t0 = 0.f, t1 = 0.f, t2 = 0.f, t3 = 0.f;
        if (gm < M) {
          if (gk + 0 < K) t0 = X[(size_t)gm * ldx + xoff + gk + 0];
          if (gk + 1 < K) t1 = X[(size_t)gm * ldx + xoff + gk + 1];
          if (gk + 2 < K) t2 = X[(size_t)gm * ldx + xoff + gk + 2];
          if (gk + 3 < K) t3 = X[(size_t)gm * ldx + xoff + gk + 3];
        }
        *(float4*)(&xs[r * LDSX + q4 * 4]) = make_float4(t0, t1, t2, t3);
      }
    }
#if ASYNC_STAGE
    asm volatile("s_wait_asynccnt 0x0" ::: "memory");
#endif
    __syncthreads();
    if (nb < N) {  // wave-uniform: EXEC all-ones inside (WMMA requirement)
      if (k0 + 32 < K && lane == 0) {
        const float* nxt = TRANSB ? &W[(size_t)nb * K + k0 + 32]
                                  : &W[(size_t)(k0 + 32) * N + nb];
        __builtin_prefetch(nxt, 0, 3);  // global_prefetch_b8
      }
      v16bf b = load_b_glb<TRANSB>(W, K, N, k0, nb, lane);
#pragma unroll
      for (int t = 0; t < GEMM_MT; ++t) {
        v16bf a = load_a_lds(xs + t * 16 * LDSX, LDSX, 0, lane);
        c[t] = wmma_bf16(a, b, c[t]);
      }
    }
    __syncthreads();
  }
  if (nb < N) {
    const int col = lane & 15, kg = lane >> 4;
    const int n = nb + col;
    const float bv = bias ? bias[n] : 0.f;
#pragma unroll
    for (int t = 0; t < GEMM_MT; ++t) {
#pragma unroll
      for (int r = 0; r < 8; ++r) {
        const int m = m0 + t * 16 + kg * 8 + r;
        float y = c[t][r] + bv;
        if (ACT == 1) y = fmaxf(y, 0.f);
        if (m < M && n < N) Y[(size_t)m * N + n] = y;
      }
    }
  }
}

// ---------------------------------------------------------------------------
// GEMM + row LayerNorm, N fixed to 256.  Block: 512 threads = 16 waves, one
// wave per 16-wide N-tile so a block owns complete rows (LN needs full rows).
// ---------------------------------------------------------------------------
template <int ACT>  // 0 = none, 1 = relu before LN
__global__ void k_gemm_ln256(const float* __restrict__ X, int ldx, int xoff,
                             const float* __restrict__ W,
                             const float* __restrict__ bias,
                             const float* __restrict__ g,
                             const float* __restrict__ be,
                             float* __restrict__ Y, int M, int K) {
  __shared__ float xs[16 * LDSX];
  __shared__ float ys[16 * LNS];
  __shared__ float s_mean[16], s_rstd[16];
  const int tid = threadIdx.x;
  const int lane = tid & 31;
  const int wave = tid >> 5;      // 0..15 = N tile
  const int m0 = blockIdx.x * 16;

  v8f c = {};
  for (int k0 = 0; k0 < K; k0 += 32) {
    for (int i = tid; i < 16 * 32; i += blockDim.x) {
      const int r = i >> 5, kk = i & 31;
      const int gk = k0 + kk, gm = m0 + r;
      float v = 0.f;
      if (gk < K && gm < M) v = X[(size_t)gm * ldx + xoff + gk];
      xs[r * LDSX + kk] = v;
    }
    __syncthreads();
    {
      v16bf a = load_a_lds(xs, LDSX, 0, lane);
      v16bf b = load_b_glb<false>(W, K, 256, k0, wave * 16, lane);
      c = wmma_bf16(a, b, c);
    }
    __syncthreads();
  }
  // park into LDS for row-wise LN
  {
    const int col = wave * 16 + (lane & 15);
    const int kg = lane >> 4;
#pragma unroll
    for (int r = 0; r < 8; ++r) {
      float y = c[r] + (bias ? bias[col] : 0.f);
      if (ACT == 1) y = fmaxf(y, 0.f);
      ys[(kg * 8 + r) * LNS + col] = y;
    }
  }
  __syncthreads();
  {  // wave w reduces row w (wave32 shfl_xor tree)
    float sum = 0.f, sq = 0.f;
    for (int i = lane; i < 256; i += 32) {
      float v = ys[wave * LNS + i];
      sum += v; sq += v * v;
    }
    for (int off = 16; off; off >>= 1) {
      sum += __shfl_xor(sum, off, 32);
      sq  += __shfl_xor(sq,  off, 32);
    }
    if (lane == 0) {
      float m = sum * (1.f / 256.f);
      float var = sq * (1.f / 256.f) - m * m;
      s_mean[wave] = m;
      s_rstd[wave] = rsqrtf(var + 1e-5f);
    }
  }
  __syncthreads();
  for (int idx = tid; idx < 16 * 256; idx += blockDim.x) {
    const int r = idx >> 8, cno = idx & 255;
    const int m = m0 + r;
    float v = (ys[r * LNS + cno] - s_mean[r]) * s_rstd[r] * g[cno] + be[cno];
    if (m < M) Y[(size_t)m * 256 + cno] = v;
  }
}

// ---------------------------------------------------------------------------
// Fused entity pipeline: e = LN(feats(M,32) @ We + be); k = e@Wk+bk,
// v = e@Wv+bv stored as bf16; per-row mask = (sum(feats) != 0).
// Block: 512 threads = 16 waves (one per N-tile of 256).
// ---------------------------------------------------------------------------
__global__ void k_ent(const float* __restrict__ feats,
                      const float* __restrict__ We, const float* __restrict__ bE,
                      const float* __restrict__ g,  const float* __restrict__ be,
                      const float* __restrict__ Wk, const float* __restrict__ bk,
                      const float* __restrict__ Wv, const float* __restrict__ bv,
                      bf16_t* __restrict__ kout, bf16_t* __restrict__ vout,
                      float* __restrict__ maskout, int M) {
  __shared__ float xs[16 * LDSX];
  __shared__ float ys[16 * LNS];
  __shared__ float s_mean[16], s_rstd[16];
  const int tid = threadIdx.x;
  const int lane = tid & 31;
  const int wave = tid >> 5;
  const int m0 = blockIdx.x * 16;

  // stage raw features (K = 32 exactly -> single WMMA K-step)
  for (int i = tid; i < 16 * 32; i += blockDim.x) {
    const int r = i >> 5, kk = i & 31;
    const int gm = m0 + r;
    xs[r * LDSX + kk] = (gm < M) ? feats[(size_t)gm * 32 + kk] : 0.f;
  }
  __syncthreads();
  if (tid < 16) {  // entity validity mask from raw features
    float s = 0.f;
    for (int k = 0; k < 32; ++k) s += xs[tid * LDSX + k];
    if (m0 + tid < M) maskout[m0 + tid] = (s != 0.f) ? 1.f : 0.f;
  }
  // embedding GEMM
  v8f ce = {};
  {
    v16bf a = load_a_lds(xs, LDSX, 0, lane);
    v16bf b = load_b_glb<false>(We, 32, 256, 0, wave * 16, lane);
    ce = wmma_bf16(a, b, ce);
  }
  {
    const int col = wave * 16 + (lane & 15);
    const int kg = lane >> 4;
#pragma unroll
    for (int r = 0; r < 8; ++r)
      ys[(kg * 8 + r) * LNS + col] = ce[r] + bE[col];
  }
  __syncthreads();
  {  // LN stats: wave w -> row w
    float sum = 0.f, sq = 0.f;
    for (int i = lane; i < 256; i += 32) {
      float v = ys[wave * LNS + i];
      sum += v; sq += v * v;
    }
    for (int off = 16; off; off >>= 1) {
      sum += __shfl_xor(sum, off, 32);
      sq  += __shfl_xor(sq,  off, 32);
    }
    if (lane == 0) {
      float m = sum * (1.f / 256.f);
      float var = sq * (1.f / 256.f) - m * m;
      s_mean[wave] = m;
      s_rstd[wave] = rsqrtf(var + 1e-5f);
    }
  }
  __syncthreads();
  for (int idx = tid; idx < 16 * 256; idx += blockDim.x) {
    const int r = idx >> 8, cno = idx & 255;
    ys[r * LNS + cno] =
        (ys[r * LNS + cno] - s_mean[r]) * s_rstd[r] * g[cno] + be[cno];
  }
  __syncthreads();
  // k / v projections straight from the normalized embedding in LDS
  v8f ck = {}, cv = {};
  for (int k0 = 0; k0 < 256; k0 += 32) {
    v16bf a   = load_a_lds(ys, LNS, k0, lane);
    v16bf bk2 = load_b_glb<false>(Wk, 256, 256, k0, wave * 16, lane);
    ck = wmma_bf16(a, bk2, ck);
    v16bf bv2 = load_b_glb<false>(Wv, 256, 256, k0, wave * 16, lane);
    cv = wmma_bf16(a, bv2, cv);
  }
  {
    const int col = wave * 16 + (lane & 15);
    const int kg = lane >> 4;
#pragma unroll
    for (int r = 0; r < 8; ++r) {
      const int m = m0 + kg * 8 + r;
      if (m < M) {
        kout[(size_t)m * 256 + col] = (__bf16)(ck[r] + bk[col]);
        vout[(size_t)m * 256 + col] = (__bf16)(cv[r] + bv[col]);
      }
    }
  }
}

// ---------------------------------------------------------------------------
// Masked multi-head attention core: one block (64 threads) per sample.
// q: (B,256) f32; k,v: (B,N,256) bf16; ctx: (B,256) f32.  NH=4, dh=64.
// ---------------------------------------------------------------------------
__global__ void k_attn(const float* __restrict__ q,
                       const bf16_t* __restrict__ kbuf,
                       const bf16_t* __restrict__ vbuf,
                       const float* __restrict__ mask,
                       float* __restrict__ ctx, int N) {
  const int b = blockIdx.x;
  const int t = threadIdx.x;  // 0..63
  const int h = t >> 4;       // head 0..3
  const int j = t & 15;
  __shared__ float sw[4][16];

  const float* qb = q + (size_t)b * 256 + h * 64;
  float s = -1e9f;
  if (j < N) {
    const bf16_t* kr = kbuf + ((size_t)b * N + j) * 256 + h * 64;
    float acc = 0.f;
#pragma unroll 8
    for (int d = 0; d < 64; ++d) acc += qb[d] * (float)kr[d];
    s = (mask[(size_t)b * N + j] != 0.f) ? acc * 0.125f : -1e9f;
  }
  sw[h][j] = s;
  __syncthreads();
  if (j == 0) {
    float mx = -1e30f;
    for (int n = 0; n < N; ++n) mx = fmaxf(mx, sw[h][n]);
    float sum = 0.f;
    for (int n = 0; n < N; ++n) {
      float e = __expf(sw[h][n] - mx);
      sw[h][n] = e; sum += e;
    }
    float inv = 1.f / sum;
    for (int n = 0; n < N; ++n) sw[h][n] *= inv;
  }
  __syncthreads();
  const bf16_t* vb = vbuf + (size_t)b * N * 256 + h * 64;
#pragma unroll
  for (int dd = 0; dd < 4; ++dd) {
    const int d = j * 4 + dd;
    float acc = 0.f;
    for (int n = 0; n < N; ++n) acc += sw[h][n] * (float)vb[(size_t)n * 256 + d];
    ctx[(size_t)b * 256 + h * 64 + d] = acc;
  }
}

// ---------------------------------------------------------------------------
// Elementwise tails
// ---------------------------------------------------------------------------
__global__ void k_concat4(const float* a, const float* b, const float* c,
                          const float* d, float* out, int B) {
  size_t i = (size_t)blockIdx.x * blockDim.x + threadIdx.x;
  if (i >= (size_t)B * 1024) return;
  size_t row = i >> 10; int col = (int)(i & 1023);
  float v;
  if (col < 256)       v = a[row * 256 + col];
  else if (col < 512)  v = b[row * 256 + (col - 256)];
  else if (col < 768)  v = c[row * 256 + (col - 512)];
  else                 v = d[row * 256 + (col - 768)];
  out[i] = v;
}

__global__ void k_concat_ctx_role(const float* ctx, const float* rp,
                                  float* out, int B) {
  size_t i = (size_t)blockIdx.x * blockDim.x + threadIdx.x;
  if (i >= (size_t)B * 320) return;
  size_t row = i / 320; int col = (int)(i % 320);
  out[i] = (col < 256) ? ctx[row * 256 + col] : rp[row * 128 + (col - 256)];
}

// GRU gates + h update; also assembles q_in = [h | role].
__global__ void k_gru(const float* __restrict__ gi, const float* __restrict__ gh,
                      const float* __restrict__ hprev, const float* __restrict__ rp,
                      float* __restrict__ h_out, float* __restrict__ q_in, int B) {
  size_t i = (size_t)blockIdx.x * blockDim.x + threadIdx.x;
  if (i >= (size_t)B * 256) return;
  const size_t b = i >> 8; const int j = (int)(i & 255);
  const float* gib = gi + b * 768;
  const float* ghb = gh + b * 768;
  float r = 1.f / (1.f + __expf(-(gib[j] + ghb[j])));
  float z = 1.f / (1.f + __expf(-(gib[256 + j] + ghb[256 + j])));
  float n = tanhf(gib[512 + j] + r * ghb[512 + j]);
  float hp = hprev[b * 256 + j];
  float h = (1.f - z) * n + z * hp;
  h_out[b * 256 + j] = h;
  q_in[b * 320 + j] = h;
  if (j < 64) q_in[b * 320 + 256 + j] = rp[b * 128 + j];
}

// ---------------------------------------------------------------------------
// Host orchestration
// ---------------------------------------------------------------------------
extern "C" void kernel_launch(void* const* d_in, const int* in_sizes, int n_in,
                              void* d_out, int out_size, void* d_ws, size_t ws_size,
                              hipStream_t stream) {
  (void)n_in; (void)out_size; (void)ws_size;
  const float* own    = (const float*)d_in[0];   // (B,1,32)
  const float* ally   = (const float*)d_in[1];   // (B,9,32)
  const float* enemy  = (const float*)d_in[2];   // (B,10,32)
  const float* hidden = (const float*)d_in[3];   // (B,256)
  const float* cap_w  = (const float*)d_in[5];
  const float* cap_b  = (const float*)d_in[6];
  const float* cap_g  = (const float*)d_in[7];
  const float* cap_be = (const float*)d_in[8];
  const float* own_w  = (const float*)d_in[9];
  const float* own_b  = (const float*)d_in[10];
  const float* al_w   = (const float*)d_in[11];
  const float* al_b   = (const float*)d_in[12];
  const float* al_g   = (const float*)d_in[13];
  const float* al_be  = (const float*)d_in[14];
  const float* en_w   = (const float*)d_in[15];
  const float* en_b   = (const float*)d_in[16];
  const float* en_g   = (const float*)d_in[17];
  const float* en_be  = (const float*)d_in[18];
  const float* aa_wq = (const float*)d_in[19]; const float* aa_bq = (const float*)d_in[20];
  const float* aa_wk = (const float*)d_in[21]; const float* aa_bk = (const float*)d_in[22];
  const float* aa_wv = (const float*)d_in[23]; const float* aa_bv = (const float*)d_in[24];
  const float* aa_wo = (const float*)d_in[25]; const float* aa_bo = (const float*)d_in[26];
  const float* ea_wq = (const float*)d_in[27]; const float* ea_bq = (const float*)d_in[28];
  const float* ea_wk = (const float*)d_in[29]; const float* ea_bk = (const float*)d_in[30];
  const float* ea_wv = (const float*)d_in[31]; const float* ea_bv = (const float*)d_in[32];
  const float* ea_wo = (const float*)d_in[33]; const float* ea_bo = (const float*)d_in[34];
  const float* cp_w  = (const float*)d_in[35]; const float* cp_b  = (const float*)d_in[36];
  const float* cp_g  = (const float*)d_in[37]; const float* cp_be = (const float*)d_in[38];
  const float* rg_w1 = (const float*)d_in[39]; const float* rg_b1 = (const float*)d_in[40];
  const float* rg_w2 = (const float*)d_in[41]; const float* rg_b2 = (const float*)d_in[42];
  const float* g_wih = (const float*)d_in[43]; const float* g_whh = (const float*)d_in[44];
  const float* g_bih = (const float*)d_in[45]; const float* g_bhh = (const float*)d_in[46];
  const float* q_w1  = (const float*)d_in[47]; const float* q_b1  = (const float*)d_in[48];
  const float* q_w2  = (const float*)d_in[49]; const float* q_b2  = (const float*)d_in[50];

  const int B  = in_sizes[3] / 256;  // 40960
  const int MA = B * 9;              // ally entity rows
  const int ME = B * 10;             // enemy entity rows

  // workspace bump allocator (256B aligned)
  char* wp = (char*)d_ws;
  auto alloc = [&](size_t bytes) -> void* {
    void* p = (void*)wp;
    wp += (bytes + 255) & ~(size_t)255;
    return p;
  };
  float*  own_e = (float*) alloc((size_t)B * 256 * 4);
  float*  cap_e = (float*) alloc((size_t)B * 256 * 4);
  bf16_t* kA    = (bf16_t*)alloc((size_t)MA * 256 * 2);
  bf16_t* vA    = (bf16_t*)alloc((size_t)MA * 256 * 2);
  bf16_t* kE    = (bf16_t*)alloc((size_t)ME * 256 * 2);
  bf16_t* vE    = (bf16_t*)alloc((size_t)ME * 256 * 2);
  float*  mskA  = (float*) alloc((size_t)MA * 4);
  float*  mskE  = (float*) alloc((size_t)ME * 4);
  float*  qA    = (float*) alloc((size_t)B * 256 * 4);
  float*  qE    = (float*) alloc((size_t)B * 256 * 4);
  float*  ctxA  = (float*) alloc((size_t)B * 256 * 4);
  float*  ctxE  = (float*) alloc((size_t)B * 256 * 4);
  float*  actx  = (float*) alloc((size_t)B * 256 * 4);
  float*  ectx  = (float*) alloc((size_t)B * 256 * 4);
  float*  cat   = (float*) alloc((size_t)B * 1024 * 4);
  float*  ctx   = (float*) alloc((size_t)B * 256 * 4);
  float*  rgh   = (float*) alloc((size_t)B * 256 * 4);
  float*  rp    = (float*) alloc((size_t)B * 128 * 4);
  float*  gin   = (float*) alloc((size_t)B * 320 * 4);
  float*  gi    = (float*) alloc((size_t)B * 768 * 4);
  float*  gh    = (float*) alloc((size_t)B * 768 * 4);
  float*  qin   = (float*) alloc((size_t)B * 320 * 4);
  float*  qh    = (float*) alloc((size_t)B * 512 * 4);

  float* q_out = (float*)d_out;                  // (B,16)
  float* h_out = (float*)d_out + (size_t)B * 16; // (B,256) == (bs,NAG,256) flat

  const int MB16  = B / 16;
  const int MB128 = B / 128;
  dim3 blk128(128), blk512(512), blk256(256), blk64(64);

  // 1) own_e = own @ own_w + own_b
  k_gemm<false, 0><<<dim3(MB128, 4), blk128, 0, stream>>>(own, 32, 0, own_w,
                                                          own_b, own_e, B, 256, 32);
  // 2) cap_e = LN(relu(own[:,25:29] @ cap_w + cap_b))
  k_gemm_ln256<1><<<dim3(MB16), blk512, 0, stream>>>(own, 32, 25, cap_w, cap_b,
                                                     cap_g, cap_be, cap_e, B, 4);
  // 3/4) fused entity embed+LN -> k,v (bf16) + masks
  k_ent<<<dim3(MA / 16), blk512, 0, stream>>>(ally, al_w, al_b, al_g, al_be,
                                              aa_wk, aa_bk, aa_wv, aa_bv,
                                              kA, vA, mskA, MA);
  k_ent<<<dim3(ME / 16), blk512, 0, stream>>>(enemy, en_w, en_b, en_g, en_be,
                                              ea_wk, ea_bk, ea_wv, ea_bv,
                                              kE, vE, mskE, ME);
  // 5/6) query projections
  k_gemm<false, 0><<<dim3(MB128, 4), blk128, 0, stream>>>(own_e, 256, 0, aa_wq,
                                                          aa_bq, qA, B, 256, 256);
  k_gemm<false, 0><<<dim3(MB128, 4), blk128, 0, stream>>>(own_e, 256, 0, ea_wq,
                                                          ea_bq, qE, B, 256, 256);
  // 7/8) masked softmax attention cores
  k_attn<<<dim3(B), blk64, 0, stream>>>(qA, kA, vA, mskA, ctxA, 9);
  k_attn<<<dim3(B), blk64, 0, stream>>>(qE, kE, vE, mskE, ctxE, 10);
  // 9/10) output projections
  k_gemm<false, 0><<<dim3(MB128, 4), blk128, 0, stream>>>(ctxA, 256, 0, aa_wo,
                                                          aa_bo, actx, B, 256, 256);
  k_gemm<false, 0><<<dim3(MB128, 4), blk128, 0, stream>>>(ctxE, 256, 0, ea_wo,
                                                          ea_bo, ectx, B, 256, 256);
  // 11) ctx_cat = [own_e | cap_e | a_ctx | e_ctx]
  k_concat4<<<dim3((unsigned)(((size_t)B * 1024 + 255) / 256)), blk256, 0,
              stream>>>(own_e, cap_e, actx, ectx, cat, B);
  // 12) ctx = LN(relu(ctx_cat @ cp_w + cp_b))
  k_gemm_ln256<1><<<dim3(MB16), blk512, 0, stream>>>(cat, 1024, 0, cp_w, cp_b,
                                                     cp_g, cp_be, ctx, B, 1024);
  // 13/14) role head
  k_gemm<false, 1><<<dim3(MB128, 4), blk128, 0, stream>>>(ctx, 256, 0, rg_w1,
                                                          rg_b1, rgh, B, 256, 256);
  k_gemm<false, 0><<<dim3(MB128, 2), blk128, 0, stream>>>(rgh, 256, 0, rg_w2,
                                                          rg_b2, rp, B, 128, 256);
  // 15) GRU input = [ctx | role]
  k_concat_ctx_role<<<dim3((unsigned)(((size_t)B * 320 + 255) / 256)), blk256,
                      0, stream>>>(ctx, rp, gin, B);
  // 16/17) GRU gate GEMMs (weights stored transposed -> TRANSB)
  k_gemm<true, 0><<<dim3(MB128, 12), blk128, 0, stream>>>(gin, 320, 0, g_wih,
                                                          g_bih, gi, B, 768, 320);
  k_gemm<true, 0><<<dim3(MB128, 12), blk128, 0, stream>>>(hidden, 256, 0, g_whh,
                                                          g_bhh, gh, B, 768, 256);
  // 18) GRU update -> h_out (also builds q_in = [h | role])
  k_gru<<<dim3((unsigned)(((size_t)B * 256 + 255) / 256)), blk256, 0, stream>>>(
      gi, gh, hidden, rp, h_out, qin, B);
  // 19/20) Q head
  k_gemm<false, 1><<<dim3(MB128, 8), blk128, 0, stream>>>(qin, 320, 0, q_w1,
                                                          q_b1, qh, B, 512, 320);
  k_gemm<false, 0><<<dim3(MB128, 1), blk128, 0, stream>>>(qh, 512, 0, q_w2,
                                                          q_b2, q_out, B, 16, 512);
}
